// Attention_28183575396503
// MI455X (gfx1250) — matrix operand order repaired and checked by reading, not compiled, not run on hardware
//
#include <hip/hip_runtime.h>
#include <hip/hip_bf16.h>

typedef _Float16 h16v __attribute__((ext_vector_type(16)));
typedef _Float16 h8v  __attribute__((ext_vector_type(8)));
typedef float    f8v  __attribute__((ext_vector_type(8)));

#define GAS __attribute__((address_space(1)))

#define WMMA_F16(A, B, C) \
  __builtin_amdgcn_wmma_f32_16x16x32_f16(false, (A), false, (B), (short)0, (C), false, false)

namespace {
constexpr int kB  = 2;
constexpr int kS  = 1024;
constexpr int kD  = 2048;
constexpr int kH  = 16;
constexpr int kDh = 128;
constexpr int kM  = kB * kS;   // 2048 token rows
}

// ---------------------------------------------------------------------------
// Stage 1: f32 -> f16 activation conversion
// ---------------------------------------------------------------------------
__global__ __launch_bounds__(256)
void cvt_f32_to_f16(const float* __restrict__ in, _Float16* __restrict__ out, int n) {
  int i = blockIdx.x * 256 + threadIdx.x;
  if (i < n) out[i] = (_Float16)in[i];
}

// ---------------------------------------------------------------------------
// Stage 2: pack int8-in-int32 weights [o][d] into WMMA B-fragment tiles.
// Tile = 32(k) x 16(n), tile-major [kt][nt]; within a tile, lane-major
// (32 lanes) x 16 f16 slots so each GEMM lane reads one contiguous 32B run.
// B packing used here (contiguous k): element (k,n) -> lane = n + 16*(k>>4), slot = k & 15.
// ---------------------------------------------------------------------------
__global__ __launch_bounds__(256)
void pack_w_kernel(const int* __restrict__ w, _Float16* __restrict__ out) {
  int idx  = blockIdx.x * 256 + threadIdx.x;  // < kD*kD
  int s    = idx & 15;
  int L    = (idx >> 4) & 31;
  int tile = idx >> 9;
  const int Nt = kD >> 4;                     // 128 n-tiles
  int kt = tile / Nt, nt = tile - kt * Nt;
  int o  = nt * 16 + (L & 15);
  int d  = kt * 32 + (L >> 4) * 16 + s;
  out[idx] = (_Float16)w[o * kD + d];         // int8 values are exact in f16
}

// ---------------------------------------------------------------------------
// Stage 3/6: C[M,N] = A[M,K] * Wpacked^T, per-output-channel scale fused.
// 4 waves (2x2), 64x64 tile per wave, 16 WMMAs per 32-deep k step.
// A tile (128x32 f16 = 8KB) is double-buffered through LDS with
// global_load_async_to_lds_b128 (ASYNCcnt); B comes from the packed layout
// as one contiguous 32B run per lane (true global_load_b128, not flat),
// with next-panel prefetch.
// ---------------------------------------------------------------------------
__global__ __launch_bounds__(128)
void gemm_wmma_f16(const _Float16* __restrict__ A, const _Float16* __restrict__ Wp,
                   const float* __restrict__ scale, float* __restrict__ C) {
  constexpr int K = kD, N = kD;
  constexpr int Nt = N >> 4;
  constexpr int KT = K / 32;
  __shared__ __align__(16) _Float16 asmem[2][128][32];   // 16KB double buffer

  const int t    = threadIdx.x;
  const int lane = t & 31;
  const int wave = t >> 5;
  const int lm = lane & 15, hi = lane >> 4;
  const int wm = wave >> 1, wn = wave & 1;
  const int mb = blockIdx.y * 128;
  const int n0 = blockIdx.x * 128 + wn * 64;

  // --- async A-tile copy: 4 x 16B chunks per thread per k-step ---
  // chunk c covers rows (c*32 + t/4), byte column (t&3)*16 of the 32-f16 row.
  const char* gbase = (const char*)(A + (size_t)(mb + (t >> 2)) * K) + (t & 3) * 16;
  const unsigned ldsbase = (unsigned)(uintptr_t)&asmem[0][0][0];

  auto issue_tile = [&](int kt, int buf) {
    const char* g = gbase + (size_t)kt * 64;             // 32 f16 per k-step
#pragma unroll
    for (int c = 0; c < 4; ++c) {
      unsigned l = ldsbase + (unsigned)buf * 8192u + (unsigned)((c * 128 + t) * 16);
      const void* gp = g + (size_t)c * 32 * K * 2;       // +32 rows
      asm volatile("global_load_async_to_lds_b128 %0, %1, off"
                   :: "v"(l), "v"(gp) : "memory");
    }
  };
  issue_tile(0, 0);

  f8v acc[4][4] = {};
  const GAS float* scg = (const GAS float*)(uintptr_t)scale;
  float sc[4];
#pragma unroll
  for (int j = 0; j < 4; ++j) sc[j] = scg[n0 + j * 16 + lm];

  // per-lane packed-B pointers (global AS) for the 4 n-subtiles at kt=0
  const GAS _Float16* bp[4];
#pragma unroll
  for (int j = 0; j < 4; ++j)
    bp[j] = (const GAS _Float16*)(uintptr_t)
            (Wp + (size_t)(((n0 >> 4) + j) * 32 + lane) * 16);

  for (int kt = 0; kt < KT; ++kt) {
    if (kt + 1 < KT) {
      issue_tile(kt + 1, (kt + 1) & 1);
      asm volatile("s_wait_asynccnt 4" ::: "memory");     // tile kt landed (own share)
    } else {
      asm volatile("s_wait_asynccnt 0" ::: "memory");
    }
    __syncthreads();                                      // all waves' copies visible

    h16v bf[4];
#pragma unroll
    for (int j = 0; j < 4; ++j) {
      union { h16v v; h8v p[2]; } u;
      u.p[0] = *(const GAS h8v*)bp[j];
      u.p[1] = *(const GAS h8v*)(bp[j] + 8);
      bf[j] = u.v;
      bp[j] += (size_t)Nt * 512;
      if (j == 0) __builtin_prefetch((const _Float16*)(uintptr_t)bp[0], 0, 1);
    }
    h16v af[4];
#pragma unroll
    for (int i = 0; i < 4; ++i) {
      // A layout: lane holds row wm*64+i*16+lm, k runs [16*hi,+16)
      const _Float16* lp = &asmem[kt & 1][wm * 64 + i * 16 + lm][0];
      union { h16v v; h8v p[2]; } u;
      u.p[0] = *(const h8v*)(lp + 16 * hi);
      u.p[1] = *(const h8v*)(lp + 16 * hi + 8);
      af[i] = u.v;
    }
#pragma unroll
    for (int i = 0; i < 4; ++i)
#pragma unroll
      for (int j = 0; j < 4; ++j)
        acc[i][j] = WMMA_F16(af[i], bf[j], acc[i][j]);
    __syncthreads();                                      // protect buffer reuse
  }
  // C layout: element (m,n) -> lane = n + 16*(m>=8), vgpr r = m&7
  GAS float* Cg = (GAS float*)(uintptr_t)C;
#pragma unroll
  for (int i = 0; i < 4; ++i)
#pragma unroll
    for (int j = 0; j < 4; ++j)
#pragma unroll
      for (int r = 0; r < 8; ++r)
        Cg[(size_t)(mb + wm * 64 + i * 16 + r + 8 * hi) * N + n0 + j * 16 + lm] =
            acc[i][j][r] * sc[j];
}

// ---------------------------------------------------------------------------
// Stage 4: full-row RMS norm + rotate-half RoPE; emit Q/K f16 [B,H,S,Dh],
// V transposed f16 [B,H,Dh,S] (so P*V B-fragments are key-contiguous).
// ---------------------------------------------------------------------------
__global__ __launch_bounds__(256)
void norm_rope_kernel(const float* __restrict__ qf, const float* __restrict__ kf,
                      const float* __restrict__ vf, const float* __restrict__ qn_w,
                      const float* __restrict__ kn_w, const float* __restrict__ cosT,
                      const float* __restrict__ sinT, _Float16* __restrict__ qh,
                      _Float16* __restrict__ kh, _Float16* __restrict__ vt) {
  __shared__ float red[256];
  __shared__ float lq[kD];
  __shared__ float lk[kD];
  const int row = blockIdx.x;              // b*S + s
  const int b = row / kS, s = row - b * kS;
  const int t = threadIdx.x;
  const float* q = qf + (size_t)row * kD;
  const float* k = kf + (size_t)row * kD;
  const float* v = vf + (size_t)row * kD;

  float sq = 0.f, sk = 0.f;
  for (int i = t; i < kD; i += 256) {
    float a = q[i], c = k[i];
    lq[i] = a; lk[i] = c;
    sq += a * a; sk += c * c;
  }
  red[t] = sq; __syncthreads();
  for (int o = 128; o > 0; o >>= 1) { if (t < o) red[t] += red[t + o]; __syncthreads(); }
  const float rq = rsqrtf(red[0] / kD + 1e-6f);
  __syncthreads();
  red[t] = sk; __syncthreads();
  for (int o = 128; o > 0; o >>= 1) { if (t < o) red[t] += red[t + o]; __syncthreads(); }
  const float rk = rsqrtf(red[0] / kD + 1e-6f);

  for (int i = t; i < kD; i += 256) {
    int hd = i & (kDh - 1), hx = i >> 7;
    float c  = cosT[s * kDh + hd];
    float sn = sinT[s * kDh + hd];
    int pi = (hd < 64) ? i + 64 : i - 64;
    float nq = lq[i] * rq * qn_w[i];
    float nk = lk[i] * rk * kn_w[i];
    float pq = lq[pi] * rq * qn_w[pi];
    float pk = lk[pi] * rk * kn_w[pi];
    float rsgn = (hd < 64) ? -1.f : 1.f;
    size_t qo = (((size_t)b * kH + hx) * kS + s) * kDh + hd;
    qh[qo] = (_Float16)(nq * c + rsgn * pq * sn);
    kh[qo] = (_Float16)(nk * c + rsgn * pk * sn);
    vt[(((size_t)b * kH + hx) * kDh + hd) * kS + s] = (_Float16)v[i];
  }
}

// ---------------------------------------------------------------------------
// Stage 5: flash attention. One wave = 16 query rows of one (b,h).
// Online softmax in exp2 domain; stats stay in the C-tile row mapping
// (m = r + 8*hi) so no cross-lane traffic for rescaling O.
// ---------------------------------------------------------------------------
__global__ __launch_bounds__(256)
void attn_kernel(const _Float16* __restrict__ qh, const _Float16* __restrict__ kh,
                 const _Float16* __restrict__ vt, const int* __restrict__ lengths,
                 _Float16* __restrict__ attn) {
  __shared__ __align__(16) _Float16 pl[8][16][32];  // per-wave P staging (1KB/wave)
  const int lane = threadIdx.x & 31;
  const int wave = threadIdx.x >> 5;
  const int lm = lane & 15, hi = lane >> 4;
  int task = blockIdx.x * 8 + wave;
  const int QB = kS / 16;
  const int qb = task % QB; task /= QB;
  const int h = task % kH;
  const int b = task / kH;
  const int len = lengths[b];
  const size_t bh = (size_t)b * kH + h;
  const GAS _Float16* Q  = (const GAS _Float16*)(uintptr_t)(qh + bh * kS * kDh);
  const GAS _Float16* Kp = (const GAS _Float16*)(uintptr_t)(kh + bh * kS * kDh);
  const GAS _Float16* V  = (const GAS _Float16*)(uintptr_t)(vt + bh * kDh * kS);
  const int q0 = qb * 16;

  h16v qa[4];
#pragma unroll
  for (int c = 0; c < 4; ++c) {
    const GAS _Float16* p = Q + (size_t)(q0 + lm) * kDh + c * 32 + 16 * hi;
    union { h16v v; h8v h[2]; } u;
    u.h[0] = *(const GAS h8v*)p; u.h[1] = *(const GAS h8v*)(p + 8);
    qa[c] = u.v;
  }
  f8v o[8] = {};
  float rmax[8], rsum[8];
#pragma unroll
  for (int r = 0; r < 8; ++r) { rmax[r] = -3.0e38f; rsum[r] = 0.f; }
  const float alpha = 0.08838834764831845f * 1.4426950408889634f;  // 1/sqrt(128)*log2(e)
  const int kend = ((len + 31) >> 5) << 5;

  for (int kb = 0; kb < kend; kb += 32) {
    f8v s1 = {}, s2 = {};
#pragma unroll
    for (int c = 0; c < 4; ++c) {
      union { h16v v; h8v h[2]; } u1, u2;
      const GAS _Float16* p1 = Kp + (size_t)(kb + lm) * kDh + c * 32 + 16 * hi;
      u1.h[0] = *(const GAS h8v*)p1; u1.h[1] = *(const GAS h8v*)(p1 + 8);
      s1 = WMMA_F16(qa[c], u1.v, s1);
      const GAS _Float16* p2 = Kp + (size_t)(kb + 16 + lm) * kDh + c * 32 + 16 * hi;
      u2.h[0] = *(const GAS h8v*)p2; u2.h[1] = *(const GAS h8v*)(p2 + 8);
      s2 = WMMA_F16(qa[c], u2.v, s2);
    }
    const float b1 = (kb + lm)      < len ? 0.f : -3.0e38f;
    const float b2 = (kb + 16 + lm) < len ? 0.f : -3.0e38f;
    float fac[8], p1v[8], p2v[8];
#pragma unroll
    for (int r = 0; r < 8; ++r) {
      float a1 = s1[r] * alpha + b1;
      float a2 = s2[r] * alpha + b2;
      float mx = fmaxf(a1, a2);
      for (int msk = 1; msk <= 8; msk <<= 1) mx = fmaxf(mx, __shfl_xor(mx, msk, 32));
      float nm = fmaxf(rmax[r], mx);
      float f  = exp2f(rmax[r] - nm);
      rmax[r] = nm;
      float e1 = exp2f(a1 - nm), e2 = exp2f(a2 - nm);
      float ts = e1 + e2;
      for (int msk = 1; msk <= 8; msk <<= 1) ts += __shfl_xor(ts, msk, 32);
      rsum[r] = rsum[r] * f + ts;
      fac[r] = f; p1v[r] = e1; p2v[r] = e2;
    }
#pragma unroll
    for (int dt = 0; dt < 8; ++dt)
#pragma unroll
      for (int r = 0; r < 8; ++r) o[dt][r] *= fac[r];
    // C-layout -> A-layout transpose of P through per-wave LDS slab
#pragma unroll
    for (int r = 0; r < 8; ++r) {
      int m = r + 8 * hi;
      pl[wave][m][lm]      = (_Float16)p1v[r];
      pl[wave][m][lm + 16] = (_Float16)p2v[r];
    }
    asm volatile("s_wait_dscnt 0" ::: "memory");
    union { h16v v; h8v h[2]; } pu;
    pu.h[0] = *(const h8v*)&pl[wave][lm][16 * hi];
    pu.h[1] = *(const h8v*)&pl[wave][lm][16 * hi + 8];
#pragma unroll
    for (int dt = 0; dt < 8; ++dt) {
      union { h16v v; h8v h[2]; } vu;
      const GAS _Float16* vp = V + (size_t)(dt * 16 + lm) * kS + kb + 16 * hi;
      vu.h[0] = *(const GAS h8v*)vp; vu.h[1] = *(const GAS h8v*)(vp + 8);
      o[dt] = WMMA_F16(pu.v, vu.v, o[dt]);
    }
  }
  float inv[8];
#pragma unroll
  for (int r = 0; r < 8; ++r) inv[r] = 1.0f / rsum[r];
  GAS _Float16* ag = (GAS _Float16*)(uintptr_t)attn;
#pragma unroll
  for (int dt = 0; dt < 8; ++dt)
#pragma unroll
    for (int r = 0; r < 8; ++r)
      ag[((size_t)b * kS + q0 + r + 8 * hi) * kD + h * kDh + dt * 16 + lm] =
          (_Float16)(o[dt][r] * inv[r]);
}

// ---------------------------------------------------------------------------
extern "C" void kernel_launch(void* const* d_in, const int* in_sizes, int n_in,
                              void* d_out, int out_size, void* d_ws, size_t ws_size,
                              hipStream_t stream) {
  (void)in_sizes; (void)n_in; (void)out_size; (void)ws_size;
  const float* x    = (const float*)d_in[0];
  const float* wq_s = (const float*)d_in[1];
  const float* wk_s = (const float*)d_in[2];
  const float* wv_s = (const float*)d_in[3];
  const float* wo_s = (const float*)d_in[4];
  const float* qn_w = (const float*)d_in[5];
  const float* kn_w = (const float*)d_in[6];
  const float* cosT = (const float*)d_in[7];
  const float* sinT = (const float*)d_in[8];
  const int*   wq   = (const int*)d_in[9];
  const int*   wk   = (const int*)d_in[10];
  const int*   wv   = (const int*)d_in[11];
  const int*   wo   = (const int*)d_in[12];
  const int*   lengths = (const int*)d_in[13];

  char* ws = (char*)d_ws;
  size_t off = 0;
  auto alloc = [&](size_t bytes) -> void* {
    void* p = ws + off;
    off += (bytes + 255) & ~(size_t)255;
    return p;
  };
  const size_t MD = (size_t)kM * kD;
  _Float16* xh  = (_Float16*)alloc(MD * 2);
  _Float16* wpq = (_Float16*)alloc((size_t)kD * kD * 2);
  _Float16* wpk = (_Float16*)alloc((size_t)kD * kD * 2);
  _Float16* wpv = (_Float16*)alloc((size_t)kD * kD * 2);
  _Float16* wpo = (_Float16*)alloc((size_t)kD * kD * 2);
  float* qf = (float*)alloc(MD * 4);
  float* kf = (float*)alloc(MD * 4);
  float* vf = (float*)alloc(MD * 4);
  _Float16* qhd = (_Float16*)alloc(MD * 2);
  _Float16* khd = (_Float16*)alloc(MD * 2);
  _Float16* vtd = (_Float16*)alloc(MD * 2);
  _Float16* attn = (_Float16*)qf;  // qf is dead after norm_rope; overlay

  const int nElem = kM * kD;       // 4,194,304
  cvt_f32_to_f16<<<nElem / 256, 256, 0, stream>>>(x, xh, nElem);
  pack_w_kernel<<<nElem / 256, 256, 0, stream>>>(wq, wpq);
  pack_w_kernel<<<nElem / 256, 256, 0, stream>>>(wk, wpk);
  pack_w_kernel<<<nElem / 256, 256, 0, stream>>>(wv, wpv);
  pack_w_kernel<<<nElem / 256, 256, 0, stream>>>(wo, wpo);

  dim3 gg(kD / 128, kM / 128);
  gemm_wmma_f16<<<gg, 128, 0, stream>>>(xh, wpq, wq_s, qf);
  gemm_wmma_f16<<<gg, 128, 0, stream>>>(xh, wpk, wk_s, kf);
  gemm_wmma_f16<<<gg, 128, 0, stream>>>(xh, wpv, wv_s, vf);

  norm_rope_kernel<<<kM, 256, 0, stream>>>(qf, kf, vf, qn_w, kn_w, cosT, sinT,
                                           qhd, khd, vtd);

  attn_kernel<<<(kB * kH * (kS / 16)) / 8, 256, 0, stream>>>(qhd, khd, vtd,
                                                             lengths, attn);

  gemm_wmma_f16<<<gg, 128, 0, stream>>>(attn, wpo, wo_s, (float*)d_out);
}
